// SelfAttention_31267361914981
// MI455X (gfx1250) — matrix operand order repaired and checked by reading, not compiled, not run on hardware
//
#include <hip/hip_runtime.h>
#include <hip/hip_bf16.h>

// ---------------------------------------------------------------------------
// Self-attention (EMBED=1024, HEADS=16, HEAD_DIM=64, N=2, L=2048) on gfx1250.
// Pipeline: f32->f16 weight convert -> per-head QKV projection (WMMA) ->
// flash attention (WMMA, 32 queries/wave, f32 online softmax, prefetch) ->
// output projection (WMMA, 32 tokens/wave).
// ---------------------------------------------------------------------------

#define EMBED 1024
#define HEADS 16
#define HDIM  64
#define NB    2
#define SEQ   2048

typedef __attribute__((ext_vector_type(16))) _Float16 v16h;
typedef __attribute__((ext_vector_type(8)))  _Float16 v8h;
typedef __attribute__((ext_vector_type(8)))  float    v8f;

static __device__ inline v8f wmma_f16(v16h a, v16h b, v8f c) {
    // D = A(16x32 f16) * B(32x16 f16) + C(16x16 f32)
    return __builtin_amdgcn_wmma_f32_16x16x32_f16(
        /*neg_a=*/false, a, /*neg_b=*/false, b,
        /*c_mod=*/(short)0, c, /*reuse_a=*/false, /*reuse_b=*/false);
}

// A-matrix layout (16-bit, 16x32): lane t (g=t/16, r=t%16) holds row M=r,
// halves 0..7  -> K = kbase + 8*g + i
// halves 8..15 -> K = kbase + 16 + 8*g + (i-8)
static __device__ inline v16h load_a16(const _Float16* __restrict__ base,
                                       int stride, int g, int r, int kbase) {
    const _Float16* p = base + (size_t)r * stride;
    v8h lo = *(const v8h*)(p + kbase + 8 * g);
    v8h hi = *(const v8h*)(p + kbase + 16 + 8 * g);
    v16h a;
#pragma unroll
    for (int i = 0; i < 8; ++i) { a[i] = lo[i]; a[i + 8] = hi[i]; }
    return a;
}

// Same A layout but converting from f32 source.
static __device__ inline v16h load_a32f(const float* __restrict__ base,
                                        int stride, int g, int r, int kbase) {
    const float* p = base + (size_t)r * stride;
    v16h a;
#pragma unroll
    for (int i = 0; i < 8; ++i) {
        a[i]     = (_Float16)p[kbase + 8 * g + i];
        a[i + 8] = (_Float16)p[kbase + 16 + 8 * g + i];
    }
    return a;
}

// B-matrix layout (16-bit, 32x16): lane t holds column N=r,
// halves 0..7 -> K = kbase + 8*g + i, halves 8..15 -> K = kbase + 16 + 8*g + (i-8).
// Source is row-major [N][K] (contiguous K).
static __device__ inline v16h load_b16(const _Float16* __restrict__ base,
                                       int stride, int g, int r, int kbase) {
    const _Float16* p = base + (size_t)r * stride + kbase + 8 * g;
    v8h lo = *(const v8h*)(p);
    v8h hi = *(const v8h*)(p + 16);
    v16h b;
#pragma unroll
    for (int i = 0; i < 8; ++i) { b[i] = lo[i]; b[i + 8] = hi[i]; }
    return b;
}

// max / sum across the 16 lanes of a half-wave (row reduction in C layout)
static __device__ inline float rowmax16(float x) {
#pragma unroll
    for (int m = 1; m < 16; m <<= 1) x = fmaxf(x, __shfl_xor(x, m, 32));
    return x;
}
static __device__ inline float rowsum16(float x) {
#pragma unroll
    for (int m = 1; m < 16; m <<= 1) x += __shfl_xor(x, m, 32);
    return x;
}

// ---------------------------------------------------------------------------
// Kernel 0: convert weights to f16
// ---------------------------------------------------------------------------
__global__ void convert_w_kernel(const float* __restrict__ Wo,
                                 const float* __restrict__ Wq,
                                 const float* __restrict__ Wk,
                                 const float* __restrict__ Wv,
                                 _Float16* __restrict__ Woh,
                                 _Float16* __restrict__ Wqh,
                                 _Float16* __restrict__ Wkh,
                                 _Float16* __restrict__ Wvh) {
    size_t i = (size_t)blockIdx.x * blockDim.x + threadIdx.x;
    if (i < (size_t)EMBED * EMBED) Woh[i] = (_Float16)Wo[i];
    if (i < (size_t)HDIM * HDIM) {
        Wqh[i] = (_Float16)Wq[i];
        Wkh[i] = (_Float16)Wk[i];
        Wvh[i] = (_Float16)Wv[i];
    }
}

// ---------------------------------------------------------------------------
// Kernel 1: per-head Q/K/V projection, out = x(16x64) @ W^T(64x64), f16 out.
// V is written transposed (d-major) so P@V B-operands load contiguously.
// ---------------------------------------------------------------------------
static __device__ inline void proj_one(const float* __restrict__ x,
                                       const _Float16* __restrict__ W,
                                       _Float16* __restrict__ out,
                                       int rowStride, int colStride,
                                       int g, int r) {
    v16h a0 = load_a32f(x, EMBED, g, r, 0);
    v16h a1 = load_a32f(x, EMBED, g, r, 32);
#pragma unroll
    for (int nb = 0; nb < HDIM; nb += 16) {
        // B[k=d_in][n=e_out] = W[e_out][d_in] (contiguous in d_in)
        v16h b0 = load_b16(W + (size_t)nb * HDIM, HDIM, g, r, 0);
        v16h b1 = load_b16(W + (size_t)nb * HDIM, HDIM, g, r, 32);
        v8f c = {};
        c = wmma_f16(a0, b0, c);
        c = wmma_f16(a1, b1, c);
#pragma unroll
        for (int v = 0; v < 8; ++v) {
            int ql = 8 * g + v;                   // local row (token)
            out[(size_t)ql * rowStride + (size_t)(nb + r) * colStride] =
                (_Float16)c[v];
        }
    }
}

__global__ __launch_bounds__(32)
void proj_kernel(const float* __restrict__ Vx, const float* __restrict__ Kx,
                 const float* __restrict__ Qx,
                 const _Float16* __restrict__ Wvh,
                 const _Float16* __restrict__ Wkh,
                 const _Float16* __restrict__ Wqh,
                 _Float16* __restrict__ Qh, _Float16* __restrict__ Kh,
                 _Float16* __restrict__ Vth) {
    const int tt = blockIdx.x;          // token tile (16 tokens)
    const int h  = blockIdx.y;
    const int n  = blockIdx.z;
    const int t  = threadIdx.x, g = t >> 4, r = t & 15;
    const int tb = tt * 16;

    const size_t xoff = ((size_t)n * SEQ + tb) * EMBED + (size_t)h * HDIM;
    const size_t hseq = ((size_t)(n * HEADS + h)) * SEQ;

    proj_one(Qx + xoff, Wqh, Qh + (hseq + tb) * HDIM, HDIM, 1, g, r);
    proj_one(Kx + xoff, Wkh, Kh + (hseq + tb) * HDIM, HDIM, 1, g, r);
    // V transposed: Vt[(n*H+h)*HDIM + d][token]
    proj_one(Vx + xoff, Wvh,
             Vth + ((size_t)(n * HEADS + h)) * HDIM * SEQ + tb, 1, SEQ, g, r);
}

// ---------------------------------------------------------------------------
// Kernel 2: flash attention; one wave handles 32 queries (2 q-tiles) so each
// K/V B-tile load feeds two WMMAs. K/V blocks one step ahead are prefetched
// (global_prefetch_b8 path).
// ---------------------------------------------------------------------------
__global__ __launch_bounds__(32)
void attn_kernel(const _Float16* __restrict__ Qh,
                 const _Float16* __restrict__ Kh,
                 const _Float16* __restrict__ Vt,
                 const int* __restrict__ mask,
                 _Float16* __restrict__ Xh) {
    const int qt = blockIdx.x;          // 32-query tile
    const int h  = blockIdx.y;
    const int n  = blockIdx.z;
    const int t  = threadIdx.x, g = t >> 4, r = t & 15;
    const int qb = qt * 32;

    const _Float16* Qbase = Qh + (((size_t)(n * HEADS + h)) * SEQ + qb) * HDIM;
    const _Float16* Kbase = Kh + ((size_t)(n * HEADS + h)) * SEQ * HDIM;
    const _Float16* Vbase = Vt + ((size_t)(n * HEADS + h)) * HDIM * SEQ;
    const int* mbase = mask + ((size_t)n * SEQ + qb) * SEQ;

    // Q tiles, A-layout: [qtile][d-chunk]
    v16h aq[2][2];
#pragma unroll
    for (int q2 = 0; q2 < 2; ++q2) {
        aq[q2][0] = load_a16(Qbase + (size_t)q2 * 16 * HDIM, HDIM, g, r, 0);
        aq[q2][1] = load_a16(Qbase + (size_t)q2 * 16 * HDIM, HDIM, g, r, 32);
    }

    v8f o[2][4];
    float m[2][8], l[2][8];
#pragma unroll
    for (int q2 = 0; q2 < 2; ++q2)
#pragma unroll
        for (int v = 0; v < 8; ++v) {
            m[q2][v] = -1e30f;
            l[q2][v] = 0.0f;
            o[q2][0][v] = 0.f; o[q2][1][v] = 0.f;
            o[q2][2][v] = 0.f; o[q2][3][v] = 0.f;
        }

    __shared__ __align__(32) _Float16 pbuf[32 * 32];  // P staging (C -> A layout)
    const float scale = 0.03125f;                     // 1/sqrt(EMBED)

    for (int kb = 0; kb < SEQ; kb += 32) {
        // ---- prefetch next K/V 32-key block (global_prefetch_b8) ----
        if (kb + 32 < SEQ) {
            __builtin_prefetch(Kbase + (size_t)(kb + 32) * HDIM + (size_t)t * 64, 0, 3);
            __builtin_prefetch(Vbase + (size_t)t * SEQ + kb + 32, 0, 3);
            __builtin_prefetch(Vbase + (size_t)(t + 32) * SEQ + kb + 32, 0, 3);
        }

        // ---- K B-tiles, shared by both q-tiles: [key-tile][d-chunk] ----
        v16h bk[2][2];
#pragma unroll
        for (int kt = 0; kt < 2; ++kt) {
            bk[kt][0] = load_b16(Kbase + (size_t)(kb + kt * 16) * HDIM, HDIM, g, r, 0);
            bk[kt][1] = load_b16(Kbase + (size_t)(kb + kt * 16) * HDIM, HDIM, g, r, 32);
        }

        __syncthreads();   // protect pbuf reads of previous iteration
#pragma unroll
        for (int q2 = 0; q2 < 2; ++q2) {
            // ---- S = Q @ K^T : two 16x16 tiles ----
            v8f s0 = {}, s1 = {};
            s0 = wmma_f16(aq[q2][0], bk[0][0], s0);
            s0 = wmma_f16(aq[q2][1], bk[0][1], s0);
            s1 = wmma_f16(aq[q2][0], bk[1][0], s1);
            s1 = wmma_f16(aq[q2][1], bk[1][1], s1);

            // ---- mask + scale (row q = qb+q2*16+8g+v, col key = kb(+16)+r) ----
            const int* mrow = mbase + (size_t)(q2 * 16) * SEQ;
#pragma unroll
            for (int v = 0; v < 8; ++v) {
                const int q = 8 * g + v;
                const int mk0 = mrow[(size_t)q * SEQ + kb + r];
                const int mk1 = mrow[(size_t)q * SEQ + kb + 16 + r];
                float e0 = (mk0 == 0) ? -1e20f : s0[v];
                float e1 = (mk1 == 0) ? -1e20f : s1[v];
                s0[v] = e0 * scale;
                s1[v] = e1 * scale;
            }

            // ---- online softmax state update ----
#pragma unroll
            for (int v = 0; v < 8; ++v) {
                float mt   = rowmax16(fmaxf(s0[v], s1[v]));
                float mnew = fmaxf(m[q2][v], mt);
                float alpha = __expf(m[q2][v] - mnew);
                m[q2][v] = mnew;
                float p0 = __expf(s0[v] - mnew);
                float p1 = __expf(s1[v] - mnew);
                s0[v] = p0; s1[v] = p1;
                l[q2][v] = l[q2][v] * alpha + rowsum16(p0 + p1);
                o[q2][0][v] *= alpha; o[q2][1][v] *= alpha;
                o[q2][2][v] *= alpha; o[q2][3][v] *= alpha;
            }

            // ---- stage P to LDS (C layout f32 -> A layout f16) ----
#pragma unroll
            for (int v = 0; v < 8; ++v) {
                const int q = q2 * 16 + 8 * g + v;
                pbuf[q * 32 + r]      = (_Float16)s0[v];
                pbuf[q * 32 + 16 + r] = (_Float16)s1[v];
            }
        }
        __syncthreads();

        // ---- read P as A-tiles ----
        v16h ap[2];
#pragma unroll
        for (int q2 = 0; q2 < 2; ++q2) {
            const int row = q2 * 16 + r;
            v8h lo = *(const v8h*)&pbuf[row * 32 + 8 * g];
            v8h hi = *(const v8h*)&pbuf[row * 32 + 16 + 8 * g];
#pragma unroll
            for (int i = 0; i < 8; ++i) { ap[q2][i] = lo[i]; ap[q2][i + 8] = hi[i]; }
        }

        // ---- O += P(16x32) @ V(32x64) : V B-tiles shared by both q-tiles ----
#pragma unroll
        for (int dt = 0; dt < 4; ++dt) {
            v16h bv = load_b16(Vbase + (size_t)dt * 16 * SEQ + kb, SEQ, g, r, 0);
            o[0][dt] = wmma_f16(ap[0], bv, o[0][dt]);
            o[1][dt] = wmma_f16(ap[1], bv, o[1][dt]);
        }
    }

    // ---- finalize: divide by l, write f16 X row-major (token, EMBED) ----
    _Float16* xrow = Xh + ((size_t)n * SEQ + qb) * EMBED + (size_t)h * HDIM;
#pragma unroll
    for (int q2 = 0; q2 < 2; ++q2)
#pragma unroll
        for (int v = 0; v < 8; ++v) {
            const int q = q2 * 16 + 8 * g + v;
            const float inv = 1.0f / l[q2][v];
            xrow[(size_t)q * EMBED + 0  + r] = (_Float16)(o[q2][0][v] * inv);
            xrow[(size_t)q * EMBED + 16 + r] = (_Float16)(o[q2][1][v] * inv);
            xrow[(size_t)q * EMBED + 32 + r] = (_Float16)(o[q2][2][v] * inv);
            xrow[(size_t)q * EMBED + 48 + r] = (_Float16)(o[q2][3][v] * inv);
        }
}

// ---------------------------------------------------------------------------
// Kernel 3: out = X(4096x1024) @ Wo^T(1024x1024) + bo, f32 output.
// 32 tokens per wave so each Wo B-tile feeds two WMMAs.
// ---------------------------------------------------------------------------
__global__ __launch_bounds__(32)
void outproj_kernel(const _Float16* __restrict__ Xh,
                    const _Float16* __restrict__ Woh,
                    const float* __restrict__ bo,
                    float* __restrict__ out) {
    const int tt = blockIdx.x;   // 32-token tile: 0..NB*SEQ/32-1
    const int nt = blockIdx.y;   // output-dim tile: 0..EMBED/16-1
    const int t  = threadIdx.x, g = t >> 4, r = t & 15;

    const _Float16* xbase = Xh + (size_t)tt * 32 * EMBED;
    const _Float16* wbase = Woh + (size_t)nt * 16 * EMBED;  // row e_out=nt*16+r

    v8f c0 = {}, c1 = {};
    for (int kb = 0; kb < EMBED; kb += 32) {
        v16h b  = load_b16(wbase, EMBED, g, r, kb);  // B[k][n] = Wo[n][k]
        v16h a0 = load_a16(xbase, EMBED, g, r, kb);
        v16h a1 = load_a16(xbase + (size_t)16 * EMBED, EMBED, g, r, kb);
        c0 = wmma_f16(a0, b, c0);
        c1 = wmma_f16(a1, b, c1);
    }
    const float bias = bo[nt * 16 + r];
    float* orow = out + (size_t)tt * 32 * EMBED + (size_t)nt * 16;
#pragma unroll
    for (int v = 0; v < 8; ++v) {
        orow[(size_t)(8 * g + v) * EMBED + r]      = c0[v] + bias;
        orow[(size_t)(16 + 8 * g + v) * EMBED + r] = c1[v] + bias;
    }
}

// ---------------------------------------------------------------------------
// Host launcher
// ---------------------------------------------------------------------------
extern "C" void kernel_launch(void* const* d_in, const int* in_sizes, int n_in,
                              void* d_out, int out_size, void* d_ws, size_t ws_size,
                              hipStream_t stream) {
    (void)in_sizes; (void)n_in; (void)out_size; (void)ws_size;

    const float* values  = (const float*)d_in[0];
    const float* keys    = (const float*)d_in[1];
    const float* queries = (const float*)d_in[2];
    const int*   mask    = (const int*)d_in[3];
    const float* Wv      = (const float*)d_in[4];
    const float* Wk      = (const float*)d_in[5];
    const float* Wq      = (const float*)d_in[6];
    const float* Wo      = (const float*)d_in[7];
    const float* bo      = (const float*)d_in[8];
    float* out = (float*)d_out;

    // Workspace layout (all 32B-aligned): Woh | Wqh | Wkh | Wvh | Qh | Kh | Vth | Xh
    char* ws = (char*)d_ws;
    _Float16* Woh = (_Float16*)ws;
    _Float16* Wqh = (_Float16*)(ws + (size_t)EMBED * EMBED * 2);
    _Float16* Wkh = Wqh + (size_t)HDIM * HDIM;
    _Float16* Wvh = Wkh + (size_t)HDIM * HDIM;
    _Float16* Qh  = Wvh + (size_t)HDIM * HDIM;
    const size_t qkvElems = (size_t)NB * HEADS * SEQ * HDIM;  // 4.19M
    _Float16* Kh  = Qh + qkvElems;
    _Float16* Vth = Kh + qkvElems;
    _Float16* Xh  = Vth + qkvElems;  // NB*SEQ*EMBED f16

    // 0) weight conversion
    {
        const int total = EMBED * EMBED;
        convert_w_kernel<<<(total + 255) / 256, 256, 0, stream>>>(
            Wo, Wq, Wk, Wv, Woh, Wqh, Wkh, Wvh);
    }
    // 1) QKV projection
    proj_kernel<<<dim3(SEQ / 16, HEADS, NB), dim3(32), 0, stream>>>(
        values, keys, queries, Wvh, Wkh, Wqh, Qh, Kh, Vth);
    // 2) flash attention (32 queries per wave)
    attn_kernel<<<dim3(SEQ / 32, HEADS, NB), dim3(32), 0, stream>>>(
        Qh, Kh, Vth, mask, Xh);
    // 3) output projection + bias (32 tokens per wave)
    outproj_kernel<<<dim3(NB * SEQ / 32, EMBED / 16), dim3(32), 0, stream>>>(
        Xh, Woh, bo, out);
}
